// KvvCov_15942918602941
// MI455X (gfx1250) — matrix operand order repaired and checked by reading, not compiled, not run on hardware
//
#include <hip/hip_runtime.h>

typedef __attribute__((ext_vector_type(16))) __bf16 v16bf;
typedef __attribute__((ext_vector_type(8)))  __bf16 v8bf;
typedef __attribute__((ext_vector_type(8)))  float  v8f;

#define BB 4
#define NN 4096
#define DD 512
#define TILE 128
#define KSTEP 32
#define KITER (DD / KSTEP)      // 16
#define LROW 40                 // halfs per LDS row: 32 data + 8 pad = 80 B (conflict-free b128)
#define LCOMP (TILE * LROW)     // halfs per staged component (128 rows)
#define LSTAGE (4 * LCOMP)      // halfs per double-buffer stage (A.hi A.lo B.hi B.lo)
#define LDS_BYTES (2 * LSTAGE * 2)  // 81920 B dynamic LDS (double buffered)

// ---------------------------------------------------------------------------
// Prep: exact f32 row norms + v, and bf16 hi/lo split of basis for split-GEMM
// ---------------------------------------------------------------------------
__global__ __launch_bounds__(256) void kvv_prep(const float* __restrict__ emb,
                                                __bf16* __restrict__ hi,
                                                __bf16* __restrict__ lo,
                                                float* __restrict__ norms,
                                                float* __restrict__ vvec) {
  int wave = blockIdx.x * (blockDim.x >> 5) + (threadIdx.x >> 5);
  int lane = threadIdx.x & 31;
  if (wave >= BB * NN) return;
  const float* row = emb + (size_t)wave * (DD + 1);
  __bf16* hrow = hi + (size_t)wave * DD;
  __bf16* lrow = lo + (size_t)wave * DD;
  float s = 0.0f;
#pragma unroll
  for (int i = 0; i < DD / 32; ++i) {
    int k = lane + (i << 5);
    float x = row[k];
    s += x * x;
    __bf16 h = (__bf16)x;              // RNE
    hrow[k] = h;
    lrow[k] = (__bf16)(x - (float)h);  // residual
  }
#pragma unroll
  for (int off = 16; off > 0; off >>= 1) s += __shfl_down(s, off, 32);
  if (lane == 0) { norms[wave] = s; vvec[wave] = row[DD]; }
}

// ---------------------------------------------------------------------------
// Async global -> LDS copy (CDNA5 GLOBAL_LOAD_ASYNC_TO_LDS_B128, ASYNCcnt)
// ---------------------------------------------------------------------------
__device__ __forceinline__ void async_b128(unsigned lds_byte_off, const void* gaddr) {
  asm volatile("global_load_async_to_lds_b128 %0, %1, off"
               :: "v"(lds_byte_off), "v"(gaddr)
               : "memory");
}

template <int N>
__device__ __forceinline__ void wait_asynccnt() {
  asm volatile("s_wait_asynccnt %0" :: "n"(N) : "memory");
}

// per-thread: 8 async B128s = this thread's share of one 128x32 bf16 tile slab
__device__ __forceinline__ void issue_step(const __bf16* src, int comp, int idx,
                                           int ks, int buf) {
#pragma unroll
  for (int j = 0; j < 8; ++j) {
    int ci = idx + (j << 6);           // 0..511 chunk id within component
    int r = ci >> 2, c4 = ci & 3;      // row 0..127, 16B-chunk 0..3
    const void* g = src + (size_t)r * DD + (size_t)ks * KSTEP + (c4 << 3);
    unsigned loff = (unsigned)((buf * LSTAGE + comp * LCOMP + r * LROW + (c4 << 3)) * 2);
    async_b128(loff, g);
  }
}

__device__ __forceinline__ v16bf frag2(const __bf16* p0, const __bf16* p1) {
  v8bf a = *(const v8bf*)p0;   // ds_load_b128
  v8bf b = *(const v8bf*)p1;   // ds_load_b128
  v16bf r;
#pragma unroll
  for (int i = 0; i < 8; ++i) { r[i] = a[i]; r[i + 8] = b[i]; }
  return r;
}

// ---------------------------------------------------------------------------
// Main: symmetric tiled split-GEMM (bf16x3) + fused RBF * vv' epilogue
// ---------------------------------------------------------------------------
__global__ __launch_bounds__(256) void kvv_gemm(const __bf16* __restrict__ hi,
                                                const __bf16* __restrict__ lo,
                                                const float* __restrict__ norms,
                                                const float* __restrict__ vvec,
                                                const float* __restrict__ ksigma,
                                                float* __restrict__ out) {
  const int ti = blockIdx.y;        // row-tile (i)
  const int tj = blockIdx.x;        // col-tile (j)
  if (tj < ti) return;              // Gram matrix: upper triangle only (block-uniform exit)
  const int b = blockIdx.z;

  extern __shared__ __align__(16) __bf16 lds[];   // 2 x (A.hi A.lo B.hi B.lo), 80 KB

  const int tid    = threadIdx.x;
  const int lane   = tid & 31;
  const int wv     = tid >> 5;          // 0..7
  const int lane15 = lane & 15;
  const int hiHalf = (lane >> 4) & 1;   // lanes 16..31 hold the "high" K half
  const int mrow0  = (wv & 1) * 64;     // wave M offset (2 wave-rows x 64)
  const int ncol0  = (wv >> 1) * 32;    // wave N offset (4 wave-cols x 32)

  // async staging: each 64-thread group owns one component (wave-uniform)
  const int comp = tid >> 6;            // 0=A.hi 1=A.lo 2=B.hi 3=B.lo
  const int idx  = tid & 63;
  const size_t rowA = ((size_t)b * NN + (size_t)ti * TILE) * DD;
  const size_t rowB = ((size_t)b * NN + (size_t)tj * TILE) * DD;
  const __bf16* src = (comp & 2) ? ((comp & 1) ? lo + rowB : hi + rowB)
                                 : ((comp & 1) ? lo + rowA : hi + rowA);

  issue_step(src, comp, idx, 0, 0);     // prologue: K-step 0 -> buffer 0

  v8f acc[4][2] = {};                   // 64 M x 32 N per wave, f32

  for (int ks = 0; ks < KITER; ++ks) {
    const int buf = ks & 1;
    if (ks + 1 < KITER) {
      // other buffer's readers finished at previous iteration's trailing barrier
      issue_step(src, comp, idx, ks + 1, buf ^ 1);
      wait_asynccnt<8>();               // in-order: retires current buffer's 8 loads
    } else {
      wait_asynccnt<0>();
    }
    __syncthreads();                    // all waves' async data visible in buf

    const __bf16* sb = lds + buf * LSTAGE;

    // B fragments (K x N = 32x16): lane holds col N=lane%16, K half by lane/16
    v16bf bh[2], bl[2];
#pragma unroll
    for (int t = 0; t < 2; ++t) {
      const __bf16* pb = &sb[2 * LCOMP + (ncol0 + t * 16 + lane15) * LROW + hiHalf * 8];
      bh[t] = frag2(pb, pb + 16);
      const __bf16* pl = &sb[3 * LCOMP + (ncol0 + t * 16 + lane15) * LROW + hiHalf * 8];
      bl[t] = frag2(pl, pl + 16);
    }
#pragma unroll
    for (int s = 0; s < 4; ++s) {
      // A fragment (M x K = 16x32): lane holds row M=lane%16; K chunks {0..7,16..23}/{8..15,24..31}
      const __bf16* pa  = &sb[0 * LCOMP + (mrow0 + s * 16 + lane15) * LROW + hiHalf * 8];
      const __bf16* pal = &sb[1 * LCOMP + (mrow0 + s * 16 + lane15) * LROW + hiHalf * 8];
      v16bf ah = frag2(pa, pa + 16);
      v16bf al = frag2(pal, pal + 16);
#pragma unroll
      for (int t = 0; t < 2; ++t) {
        acc[s][t] = __builtin_amdgcn_wmma_f32_16x16x32_bf16(false, ah, false, bh[t],
                                                            (short)0, acc[s][t], false, false);
        acc[s][t] = __builtin_amdgcn_wmma_f32_16x16x32_bf16(false, ah, false, bl[t],
                                                            (short)0, acc[s][t], false, false);
        acc[s][t] = __builtin_amdgcn_wmma_f32_16x16x32_bf16(false, al, false, bh[t],
                                                            (short)0, acc[s][t], false, false);
      }
    }
    __syncthreads();                    // buf reads done; safe to overwrite next iter
  }

  // Fused epilogue: rbf(dists) * v v'
  const float coef = -0.5f * __expf(-2.0f * ksigma[0]);  // -0.5 / exp(sigma)^2
  const float* nb = norms + (size_t)b * NN;
  const float* vb = vvec + (size_t)b * NN;
  float* ob = out + (size_t)b * NN * NN;
#pragma unroll
  for (int s = 0; s < 4; ++s) {
#pragma unroll
    for (int t = 0; t < 2; ++t) {
      int jg = tj * TILE + ncol0 + t * 16 + lane15;
      float njv = nb[jg], vjv = vb[jg];
#pragma unroll
      for (int r = 0; r < 8; ++r) {
        int ig = ti * TILE + mrow0 + s * 16 + hiHalf * 8 + r;   // C layout: VGPR r -> M=r(+8)
        float d   = nb[ig] + njv - 2.0f * acc[s][t][r];
        float val = __expf(coef * d) * vb[ig] * vjv;
        ob[(size_t)ig * NN + jg] = val;
        if (ti != tj) ob[(size_t)jg * NN + ig] = val;            // mirror block
      }
    }
  }
}

// ---------------------------------------------------------------------------
extern "C" void kernel_launch(void* const* d_in, const int* in_sizes, int n_in,
                              void* d_out, int out_size, void* d_ws, size_t ws_size,
                              hipStream_t stream) {
  const float* emb  = (const float*)d_in[0];
  const float* ksig = (const float*)d_in[1];
  // d_in[2] = num_basis_dim (512, compile-time constant here)
  float* out = (float*)d_out;
  char*  ws  = (char*)d_ws;

  const size_t bfbytes = (size_t)BB * NN * DD * sizeof(__bf16);  // 16.8 MB each
  __bf16* hi    = (__bf16*)ws;
  __bf16* lo    = (__bf16*)(ws + bfbytes);
  float*  norms = (float*)(ws + 2 * bfbytes);
  float*  vvec  = (float*)(ws + 2 * bfbytes + (size_t)BB * NN * sizeof(float));

  kvv_prep<<<(BB * NN) / 8, 256, 0, stream>>>(emb, hi, lo, norms, vvec);

  dim3 grid(NN / TILE, NN / TILE, BB);
  kvv_gemm<<<grid, 256, LDS_BYTES, stream>>>(hi, lo, norms, vvec, ksig, out);

  (void)in_sizes; (void)n_in; (void)out_size; (void)ws_size;
}
